// NeighborAggregator_36455682408725
// MI455X (gfx1250) — compile-verified, matrix operand order checked by reading, not run on hardware
//
#include <hip/hip_runtime.h>
#include <hip/hip_bf16.h>

// ---------------------------------------------------------------------------
// GNN neighbor aggregation, bf16 WMMA path for gfx1250 (MI455X).
//   agg/deg/h_bf16 live in d_ws; weights live pre-swizzled (WMMA B-fragment
//   order) in LDS; per-edge-tile 16x128 intermediates stage through LDS to
//   perform the C-layout -> A-layout transpose between the two MLP layers.
// ---------------------------------------------------------------------------

#define DIM 128

typedef __attribute__((ext_vector_type(16))) __bf16 v16bf;
typedef __attribute__((ext_vector_type(8)))  float  v8f;

union FragA {
    v16bf  v;
    uint4  q[2];
    __bf16 e[16];
};

// K offset inside a 32-wide K step for the 16-bit WMMA operand layout:
//   lanes 0-15 (g=0): elements 0-7 -> K 0-7,  elements 8-15 -> K 16-23
//   lanes 16-31 (g=1): elements 0-7 -> K 8-15, elements 8-15 -> K 24-31
__device__ __forceinline__ int koff(int g, int j) {
    return (j < 8) ? (8 * g + j) : (16 + 8 * g + (j - 8));
}

// Convert f32 weight matrix W[k][n] (out-dim 128) to bf16, stored so that the
// B fragment for (nt, kt) is 16 contiguous bf16 per lane (two ds_load_b128).
// dst index = (((nt*KT + kt)*32) + lane)*16 + j ; value = W[kt*32+koff][nt*16+n]
__device__ __forceinline__ void fill_swizzled(const float* __restrict__ W,
                                              __bf16* dst, int KT,
                                              int tid, int nthreads) {
    const int total = 8 * KT * 32 * 16;
    for (int idx = tid; idx < total; idx += nthreads) {
        int j    = idx & 15;
        int lane = (idx >> 4) & 31;
        int rest = idx >> 9;
        int kt   = rest % KT;
        int nt   = rest / KT;
        int g    = lane >> 4;
        int n    = lane & 15;
        int k    = kt * 32 + koff(g, j);
        dst[idx] = (__bf16)W[k * DIM + nt * 16 + n];
    }
}

// ---------------------------------------------------------------------------
// Phase 0: zero agg/deg, convert h -> bf16 (workspace is not re-zeroed by the
// harness between replays, so this must run every launch).
// ---------------------------------------------------------------------------
__global__ void init_kernel(const float* __restrict__ h, __bf16* __restrict__ hbf,
                            float* __restrict__ agg, float* __restrict__ deg,
                            long long total, int N) {
    long long i      = (long long)blockIdx.x * blockDim.x + threadIdx.x;
    long long stride = (long long)gridDim.x * blockDim.x;
    for (; i < total; i += stride) {
        hbf[i] = (__bf16)h[i];
        agg[i] = 0.0f;
        if (i < N) deg[i] = 0.0f;
    }
}

// ---------------------------------------------------------------------------
// Phase 1: per-edge message MLP + scatter-add.  One wave per 16-edge tile.
// ---------------------------------------------------------------------------
__global__ void __launch_bounds__(256)
edge_msg_kernel(const int* __restrict__ erow, const int* __restrict__ ecol,
                const __bf16* __restrict__ hbf,
                const float* __restrict__ W1, const float* __restrict__ b1,
                const float* __restrict__ W2, const float* __restrict__ b2,
                float* __restrict__ agg, float* __restrict__ deg,
                int E, int num_tiles) {
    extern __shared__ __align__(16) char smem[];
    __bf16* W1s      = (__bf16*)smem;        // 8*4*32*16 = 16384 bf16
    __bf16* W2s      = W1s + 16384;          // 16384 bf16
    __bf16* stageAll = W2s + 16384;          // 8 waves * 16*128 bf16

    const int tid = threadIdx.x;
    fill_swizzled(W1, W1s, 4, tid, blockDim.x);
    fill_swizzled(W2, W2s, 4, tid, blockDim.x);
    __syncthreads();

    const int lane = tid & 31;
    const int wave = tid >> 5;
    const int g    = lane >> 4;
    const int nl   = lane & 15;
    __bf16* stage  = stageAll + wave * (16 * DIM);

    const int wavesPerBlk = blockDim.x >> 5;
    const int nwaves      = gridDim.x * wavesPerBlk;

    for (int tile = blockIdx.x * wavesPerBlk + wave; tile < num_tiles; tile += nwaves) {
        const int e0 = tile * 16;

        // hint L2 for next tile's indices -> global_prefetch_b8
        const int tnext = tile + nwaves;
        if (tnext < num_tiles) {
            __builtin_prefetch(&ecol[tnext * 16], 0, 1);
            __builtin_prefetch(&erow[tnext * 16], 0, 1);
        }

        int eA = e0 + nl;
        if (eA >= E) eA = E - 1;                 // padded lanes: garbage, never scattered
        const int src = ecol[eA];
        const __bf16* xrow = hbf + (size_t)src * DIM;

        // layer-1 A fragments: gather 2x16B per K step from h_bf16[src]
        FragA a[4];
#pragma unroll
        for (int kt = 0; kt < 4; ++kt) {
            a[kt].q[0] = *(const uint4*)(xrow + kt * 32 + 8 * g);
            a[kt].q[1] = *(const uint4*)(xrow + kt * 32 + 8 * g + 16);
        }

        // layer 1: m1 = relu(x @ W1 + b1), staged to LDS as bf16 16x128
#pragma unroll
        for (int nt = 0; nt < 8; ++nt) {
            const float bias = b1[nt * 16 + nl];
            v8f c;
#pragma unroll
            for (int r = 0; r < 8; ++r) c[r] = bias;
#pragma unroll
            for (int kt = 0; kt < 4; ++kt) {
                FragA bm;
                const __bf16* bp = W1s + (((nt << 2) + kt) * 32 + lane) * 16;
                bm.q[0] = *(const uint4*)bp;
                bm.q[1] = *(const uint4*)(bp + 8);
                c = __builtin_amdgcn_wmma_f32_16x16x32_bf16(
                        false, a[kt].v, false, bm.v, (short)0, c, false, false);
            }
#pragma unroll
            for (int r = 0; r < 8; ++r) {
                float v = c[r] > 0.0f ? c[r] : 0.0f;
                stage[(r + 8 * g) * DIM + nt * 16 + nl] = (__bf16)v;  // C -> row-major
            }
        }

        // layer-2 A fragments re-read from LDS in A layout (does the transpose)
        FragA a2[4];
#pragma unroll
        for (int kt = 0; kt < 4; ++kt) {
            const __bf16* sp = stage + nl * DIM + kt * 32 + 8 * g;
            a2[kt].q[0] = *(const uint4*)sp;
            a2[kt].q[1] = *(const uint4*)(sp + 16);
        }

        int dstn[8];
#pragma unroll
        for (int r = 0; r < 8; ++r) {
            const int e = e0 + r + 8 * g;
            dstn[r] = erow[(e < E) ? e : 0];
        }

        // layer 2: m = relu(m1 @ W2 + b2), scatter-add into agg[row]
#pragma unroll
        for (int nt = 0; nt < 8; ++nt) {
            const float bias = b2[nt * 16 + nl];
            v8f c;
#pragma unroll
            for (int r = 0; r < 8; ++r) c[r] = bias;
#pragma unroll
            for (int kt = 0; kt < 4; ++kt) {
                FragA bm;
                const __bf16* bp = W2s + (((nt << 2) + kt) * 32 + lane) * 16;
                bm.q[0] = *(const uint4*)bp;
                bm.q[1] = *(const uint4*)(bp + 8);
                c = __builtin_amdgcn_wmma_f32_16x16x32_bf16(
                        false, a2[kt].v, false, bm.v, (short)0, c, false, false);
            }
#pragma unroll
            for (int r = 0; r < 8; ++r) {
                const int e = e0 + r + 8 * g;
                if (e < E) {
                    float v = c[r] > 0.0f ? c[r] : 0.0f;
                    __hip_atomic_fetch_add(&agg[(size_t)dstn[r] * DIM + nt * 16 + nl], v,
                                           __ATOMIC_RELAXED, __HIP_MEMORY_SCOPE_AGENT);
                }
            }
        }

        // degree counts: one lane per edge in the tile
        if (lane < 16 && (e0 + lane) < E) {
            __hip_atomic_fetch_add(&deg[erow[e0 + lane]], 1.0f,
                                   __ATOMIC_RELAXED, __HIP_MEMORY_SCOPE_AGENT);
        }
    }
}

// ---------------------------------------------------------------------------
// Phase 2: node update MLP on z = [h, agg/max(deg,1)]  (K = 256).
// ---------------------------------------------------------------------------
__global__ void __launch_bounds__(256)
node_update_kernel(const float* __restrict__ h,
                   const float* __restrict__ agg, const float* __restrict__ deg,
                   const float* __restrict__ U1, const float* __restrict__ c1,
                   const float* __restrict__ U2, const float* __restrict__ c2,
                   float* __restrict__ out, int N, int num_tiles) {
    extern __shared__ __align__(16) char smem[];
    __bf16* U1s      = (__bf16*)smem;        // 8*8*32*16 = 32768 bf16
    __bf16* U2s      = U1s + 32768;          // 16384 bf16
    __bf16* stageAll = U2s + 16384;          // 8 waves * 16*128 bf16

    const int tid = threadIdx.x;
    fill_swizzled(U1, U1s, 8, tid, blockDim.x);
    fill_swizzled(U2, U2s, 4, tid, blockDim.x);
    __syncthreads();

    const int lane = tid & 31;
    const int wave = tid >> 5;
    const int g    = lane >> 4;
    const int nl   = lane & 15;
    __bf16* stage  = stageAll + wave * (16 * DIM);

    const int wavesPerBlk = blockDim.x >> 5;
    const int nwaves      = gridDim.x * wavesPerBlk;

    for (int tile = blockIdx.x * wavesPerBlk + wave; tile < num_tiles; tile += nwaves) {
        int node = tile * 16 + nl;
        if (node >= N) node = N - 1;             // padded lanes: store-guarded below
        float d = deg[node];
        if (d < 1.0f) d = 1.0f;
        const float invd = 1.0f / d;
        const float* hrow = h   + (size_t)node * DIM;
        const float* arow = agg + (size_t)node * DIM;

        // A fragments over K=256: first 128 from h, second 128 from agg*invd
        FragA a[8];
#pragma unroll
        for (int kt = 0; kt < 8; ++kt) {
            const float* src = (kt < 4) ? (hrow + kt * 32) : (arow + (kt - 4) * 32);
            const float  s   = (kt < 4) ? 1.0f : invd;
            float tmp[16];
            *(float4*)(tmp + 0)  = *(const float4*)(src + 8 * g);
            *(float4*)(tmp + 4)  = *(const float4*)(src + 8 * g + 4);
            *(float4*)(tmp + 8)  = *(const float4*)(src + 8 * g + 16);
            *(float4*)(tmp + 12) = *(const float4*)(src + 8 * g + 20);
#pragma unroll
            for (int j = 0; j < 16; ++j) a[kt].e[j] = (__bf16)(tmp[j] * s);
        }

        // layer 1: relu(z @ U1 + c1) -> stage
#pragma unroll
        for (int nt = 0; nt < 8; ++nt) {
            const float bias = c1[nt * 16 + nl];
            v8f c;
#pragma unroll
            for (int r = 0; r < 8; ++r) c[r] = bias;
#pragma unroll
            for (int kt = 0; kt < 8; ++kt) {
                FragA bm;
                const __bf16* bp = U1s + (((nt << 3) + kt) * 32 + lane) * 16;
                bm.q[0] = *(const uint4*)bp;
                bm.q[1] = *(const uint4*)(bp + 8);
                c = __builtin_amdgcn_wmma_f32_16x16x32_bf16(
                        false, a[kt].v, false, bm.v, (short)0, c, false, false);
            }
#pragma unroll
            for (int r = 0; r < 8; ++r) {
                float v = c[r] > 0.0f ? c[r] : 0.0f;
                stage[(r + 8 * g) * DIM + nt * 16 + nl] = (__bf16)v;
            }
        }

        FragA a2[4];
#pragma unroll
        for (int kt = 0; kt < 4; ++kt) {
            const __bf16* sp = stage + nl * DIM + kt * 32 + 8 * g;
            a2[kt].q[0] = *(const uint4*)sp;
            a2[kt].q[1] = *(const uint4*)(sp + 16);
        }

        // layer 2: z1 @ U2 + c2 -> out (no relu)
#pragma unroll
        for (int nt = 0; nt < 8; ++nt) {
            const float bias = c2[nt * 16 + nl];
            v8f c;
#pragma unroll
            for (int r = 0; r < 8; ++r) c[r] = bias;
#pragma unroll
            for (int kt = 0; kt < 4; ++kt) {
                FragA bm;
                const __bf16* bp = U2s + (((nt << 2) + kt) * 32 + lane) * 16;
                bm.q[0] = *(const uint4*)bp;
                bm.q[1] = *(const uint4*)(bp + 8);
                c = __builtin_amdgcn_wmma_f32_16x16x32_bf16(
                        false, a2[kt].v, false, bm.v, (short)0, c, false, false);
            }
#pragma unroll
            for (int r = 0; r < 8; ++r) {
                const int nm = tile * 16 + r + 8 * g;
                if (nm < N) out[(size_t)nm * DIM + nt * 16 + nl] = c[r];
            }
        }
    }
}

// ---------------------------------------------------------------------------
extern "C" void kernel_launch(void* const* d_in, const int* in_sizes, int n_in,
                              void* d_out, int out_size, void* d_ws, size_t ws_size,
                              hipStream_t stream) {
    const float* h  = (const float*)d_in[0];
    const int* eidx = (const int*)d_in[1];     // [2, E] int32 (jax x64 disabled)
    const float* W1 = (const float*)d_in[2];
    const float* b1 = (const float*)d_in[3];
    const float* W2 = (const float*)d_in[4];
    const float* b2 = (const float*)d_in[5];
    const float* U1 = (const float*)d_in[6];
    const float* c1 = (const float*)d_in[7];
    const float* U2 = (const float*)d_in[8];
    const float* c2 = (const float*)d_in[9];

    const int N = in_sizes[0] / DIM;
    const int E = in_sizes[1] / 2;
    const int* erow = eidx;
    const int* ecol = eidx + E;

    // workspace carve-up: agg [N*128 f32] | deg [N f32, padded] | h_bf16 [N*128]
    char* ws = (char*)d_ws;
    float* agg = (float*)ws;
    size_t off = (size_t)N * DIM * sizeof(float);
    float* deg = (float*)(ws + off);
    off += (((size_t)N * sizeof(float)) + 255) & ~(size_t)255;
    __bf16* hbf = (__bf16*)(ws + off);
    (void)ws_size; (void)n_in; (void)out_size;

    init_kernel<<<2048, 256, 0, stream>>>(h, hbf, agg, deg, (long long)N * DIM, N);

    const int etiles = (E + 15) / 16;
    const size_t smem_e = (size_t)(16384 + 16384 + 8 * 16 * DIM) * sizeof(__bf16); // 96 KB
    edge_msg_kernel<<<512, 256, smem_e, stream>>>(erow, ecol, hbf, W1, b1, W2, b2,
                                                  agg, deg, E, etiles);

    const int ntiles = (N + 15) / 16;
    const size_t smem_n = (size_t)(32768 + 16384 + 8 * 16 * DIM) * sizeof(__bf16); // 128 KB
    node_update_kernel<<<256, 256, smem_n, stream>>>(h, agg, deg, U1, c1, U2, c2,
                                                     (float*)d_out, N, ntiles);
}
